// MHA_GQA_6382321402211
// MI455X (gfx1250) — matrix operand order repaired and checked by reading, not compiled, not run on hardware
//
#include <hip/hip_runtime.h>
#include <hip/hip_bf16.h>
#include <stdint.h>

#define D_MODEL   2048
#define N_HEADS   16
#define N_KV      4
#define HEAD_DIM  128
#define T_SEQ     4096
#define SWA       1024
#define SINKS     4

typedef __attribute__((ext_vector_type(16))) __bf16 v16bf;
typedef __attribute__((ext_vector_type(8)))  float  v8f;

struct alignas(16) U128 { uint32_t x[4]; };
struct FragBf { v16bf v; };   // 16 bf16 = 32 bytes = 8 VGPRs

__device__ __forceinline__ void ld2x128(FragBf& f, const uint16_t* p0, const uint16_t* p1) {
    U128* d = reinterpret_cast<U128*>(&f);
    d[0] = *reinterpret_cast<const U128*>(p0);
    d[1] = *reinterpret_cast<const U128*>(p1);
}

__device__ __forceinline__ v8f wmma_bf16(const v16bf& a, const v16bf& b, const v8f& c) {
    // D = A(16x32 bf16) x B(32x16 bf16) + C(16x16 f32), wave32
    return __builtin_amdgcn_wmma_f32_16x16x32_bf16(false, a, false, b, (short)0, c, false, false);
}

__device__ __forceinline__ uint16_t f2bf(float f) {
    uint32_t u = __float_as_uint(f);
    u += 0x7fffu + ((u >> 16) & 1u);      // round-to-nearest-even
    return (uint16_t)(u >> 16);
}
__device__ __forceinline__ float bf2f(uint16_t h) {
    return __uint_as_float(((uint32_t)h) << 16);
}

// ---------------------------------------------------------------- conversion
__global__ void f32_to_bf16_kernel(const float* __restrict__ in,
                                   uint16_t* __restrict__ out, int n) {
    int i = blockIdx.x * blockDim.x + threadIdx.x;
    if (i < n) out[i] = f2bf(in[i]);
}

// ---------------------------------------------------------------- GEMM: D = X(MxK) * W(NxK)^T
// 64x64 tile per wave, K-chunks of 32, bf16 WMMA with f32 accumulate.
// STORE: 0 = bf16 row-major (ld=N), 1 = bf16 transposed Out[n*M+m], 2 = f32 row-major
template<int STORE>
__global__ __launch_bounds__(256) void gemm_wmma_kernel(
    const uint16_t* __restrict__ X, const uint16_t* __restrict__ W,
    void* __restrict__ OutP, int M, int N, int K)
{
    const int lane = threadIdx.x & 31;
    const int wid  = blockIdx.x * (blockDim.x >> 5) + (threadIdx.x >> 5);
    const int ntile = N >> 6;
    const int mt = wid / ntile;
    const int nt = wid - mt * ntile;
    const int m0 = mt << 6, n0 = nt << 6;
    if (m0 >= M) return;
    const int lh  = lane >> 4;      // half-wave select
    const int l15 = lane & 15;

    // A frag (16x32): lanes 0-15 row m=l15 hold K {0..7,16..23}; lanes 16-31 hold K {8..15,24..31}
    // B frag (32x16): lanes 0-15 col n=l15 hold K {0..7,16..23}; lanes 16-31 hold K {8..15,24..31}
    const uint16_t* xa[4];
    const uint16_t* wb[4];
#pragma unroll
    for (int i = 0; i < 4; ++i) {
        xa[i] = X + (size_t)(m0 + 16 * i + l15) * K + (lh ? 8 : 0);
        wb[i] = W + (size_t)(n0 + 16 * i + l15) * K + (lh ? 8 : 0);
    }

    const v8f zero8 = {0.f, 0.f, 0.f, 0.f, 0.f, 0.f, 0.f, 0.f};
    v8f c[16];
#pragma unroll
    for (int i = 0; i < 16; ++i) c[i] = zero8;

    for (int kc = 0; kc < K; kc += 32) {
        FragBf a[4], b[4];
#pragma unroll
        for (int i = 0; i < 4; ++i) {
            ld2x128(a[i], xa[i] + kc, xa[i] + kc + 16);
            ld2x128(b[i], wb[i] + kc, wb[i] + kc + 16);
        }
#pragma unroll
        for (int im = 0; im < 4; ++im)
#pragma unroll
            for (int in = 0; in < 4; ++in)
                c[im * 4 + in] = wmma_bf16(a[im].v, b[in].v, c[im * 4 + in]);
    }

    // C layout: lanes 0-15 elem r -> (m0+16*im+r, n0+16*in+l15); lanes 16-31 -> row +8
    if constexpr (STORE == 1) {
        uint16_t* Out = (uint16_t*)OutP;               // Out[n * M + m]
#pragma unroll
        for (int im = 0; im < 4; ++im)
#pragma unroll
            for (int in = 0; in < 4; ++in) {
                U128 t;
                uint16_t* tu = reinterpret_cast<uint16_t*>(&t);
#pragma unroll
                for (int r = 0; r < 8; ++r) tu[r] = f2bf(c[im * 4 + in][r]);
                uint16_t* p = Out + (size_t)(n0 + 16 * in + l15) * M + (m0 + 16 * im + lh * 8);
                *reinterpret_cast<U128*>(p) = t;       // 8 contiguous bf16
            }
    } else if constexpr (STORE == 2) {
        float* Out = (float*)OutP;
#pragma unroll
        for (int im = 0; im < 4; ++im)
#pragma unroll
            for (int in = 0; in < 4; ++in)
#pragma unroll
                for (int r = 0; r < 8; ++r)
                    Out[(size_t)(m0 + 16 * im + lh * 8 + r) * N + n0 + 16 * in + l15] =
                        c[im * 4 + in][r];
    } else {
        uint16_t* Out = (uint16_t*)OutP;
#pragma unroll
        for (int im = 0; im < 4; ++im)
#pragma unroll
            for (int in = 0; in < 4; ++in)
#pragma unroll
                for (int r = 0; r < 8; ++r)
                    Out[(size_t)(m0 + 16 * im + lh * 8 + r) * N + n0 + 16 * in + l15] =
                        f2bf(c[im * 4 + in][r]);
    }
}

// ---------------------------------------------------------------- RoPE + QK RMSNorm (in place, bf16)
// One wave per (token, head) row of 128. Norm is applied AFTER rotation (matches reference).
__global__ __launch_bounds__(256) void rope_qknorm_kernel(
    uint16_t* __restrict__ buf, const float* __restrict__ w, int nheads)
{
    const int lane = threadIdx.x & 31;
    const int rid  = blockIdx.x * (blockDim.x >> 5) + (threadIdx.x >> 5);
    if (rid >= T_SEQ * nheads) return;
    const int t = rid / nheads;
    uint16_t* row = buf + (size_t)rid * HEAD_DIM;

    const float L2_10K_64 = 0.20762050593046054f;   // log2(10000)/64
    float o[4];
    float ss = 0.f;
#pragma unroll
    for (int p = 0; p < 2; ++p) {
        const int i = lane + 32 * p;                 // i in [0,64)
        float x1 = bf2f(row[i]);
        float x2 = bf2f(row[i + 64]);
        float fr = (float)t * exp2f(-(float)i * L2_10K_64);
        float cs = __cosf(fr), sn = __sinf(fr);
        float lo = x1 * cs - x2 * sn;
        float hi = x1 * sn + x2 * cs;
        o[2 * p] = lo; o[2 * p + 1] = hi;
        ss += lo * lo + hi * hi;
    }
#pragma unroll
    for (int off = 16; off >= 1; off >>= 1) ss += __shfl_xor(ss, off, 32);
    const float rinv = rsqrtf(ss * (1.0f / HEAD_DIM) + 1e-6f);
#pragma unroll
    for (int p = 0; p < 2; ++p) {
        const int i = lane + 32 * p;
        row[i]      = f2bf(o[2 * p]     * rinv * w[i]);
        row[i + 64] = f2bf(o[2 * p + 1] * rinv * w[i + 64]);
    }
}

// ---------------------------------------------------------------- Flash-style SWA+sink attention
// One wave = one (head, 16-query) tile. 32-key blocks. 16 WMMAs per key block.
// K is read in natural (t, d) layout; V is read from the transposed (d, t) buffer.
__global__ __launch_bounds__(256) void swa_attn_kernel(
    const uint16_t* __restrict__ qb, const uint16_t* __restrict__ kb,
    const uint16_t* __restrict__ vt, uint16_t* __restrict__ yb)
{
    __shared__ __align__(16) uint16_t plds[8][16 * 72];   // padded stride 72 -> 16B aligned rows
    const int lane = threadIdx.x & 31;
    const int wv   = threadIdx.x >> 5;
    const int task = blockIdx.x * 8 + wv;
    const int h    = task >> 8;                  // 256 q-tiles per head
    const int q0   = (task & 255) << 4;
    const int kvh  = h >> 2;                     // GQA: 4 Q heads per KV head
    const int lh   = lane >> 4, l15 = lane & 15;
    const int rowoff = lh * 8;
    const float SCALE = 0.08838834764831845f;    // 1/sqrt(128)

    // Q A-fragments for the 4 K-chunks of head_dim
    FragBf qf[4];
    {
        const uint16_t* qp = qb + (size_t)(q0 + l15) * D_MODEL + h * HEAD_DIM + (lh ? 8 : 0);
#pragma unroll
        for (int c = 0; c < 4; ++c) ld2x128(qf[c], qp + 32 * c, qp + 32 * c + 16);
    }

    const v8f zero8 = {0.f, 0.f, 0.f, 0.f, 0.f, 0.f, 0.f, 0.f};
    v8f o[8];
    float mrow[8], lrow[8];
#pragma unroll
    for (int i = 0; i < 8; ++i) { o[i] = zero8; mrow[i] = -1e30f; lrow[i] = 0.f; }

    const int kw0 = (q0 > (SWA - 1)) ? (((q0 - (SWA - 1)) >> 5) << 5) : 0;
    const uint16_t* vh = vt + (size_t)kvh * HEAD_DIM * T_SEQ;

    for (int k0 = 0; k0 <= q0 + 15; k0 += 32) {
        if (k0 == 32 && kw0 > 32) k0 = kw0;      // skip fully-masked middle (sinks are in block 0)

        // ---- S = Q * K^T for 32 keys (two 16x16 tiles) ----
        v8f s0 = zero8, s1 = zero8;
        {
            const uint16_t* kp0 = kb + (size_t)(k0 + l15) * (N_KV * HEAD_DIM)
                                  + kvh * HEAD_DIM + lh * 8;
            const uint16_t* kp1 = kp0 + (size_t)16 * (N_KV * HEAD_DIM);
#pragma unroll
            for (int c = 0; c < 4; ++c) {
                FragBf kf0, kf1;
                ld2x128(kf0, kp0 + 32 * c, kp0 + 32 * c + 16);
                ld2x128(kf1, kp1 + 32 * c, kp1 + 32 * c + 16);
                s0 = wmma_bf16(qf[c].v, kf0.v, s0);
                s1 = wmma_bf16(qf[c].v, kf1.v, s1);
            }
        }

        // ---- mask + online softmax (row reductions within 16-lane groups) ----
        float ar[8];
#pragma unroll
        for (int r = 0; r < 8; ++r) {
            const int i  = q0 + rowoff + r;
            const int ka = k0 + l15;
            const int kbcol = k0 + 16 + l15;
            const bool ok0 = (ka <= i)    && (((i - ka)    < SWA) || (ka    < SINKS));
            const bool ok1 = (kbcol <= i) && (((i - kbcol) < SWA) || (kbcol < SINKS));
            float v0 = ok0 ? s0[r] * SCALE : -1e30f;
            float v1 = ok1 ? s1[r] * SCALE : -1e30f;
            float mx = fmaxf(v0, v1);
#pragma unroll
            for (int off = 8; off >= 1; off >>= 1) mx = fmaxf(mx, __shfl_xor(mx, off, 32));
            const float mn = fmaxf(mrow[r], mx);
            ar[r]   = __expf(mrow[r] - mn);
            mrow[r] = mn;
            const float p0 = __expf(v0 - mn);
            const float p1 = __expf(v1 - mn);
            float rs = p0 + p1;
#pragma unroll
            for (int off = 8; off >= 1; off >>= 1) rs += __shfl_xor(rs, off, 32);
            lrow[r] = lrow[r] * ar[r] + rs;
            // stage P (C layout -> LDS) for A-layout reload
            plds[wv][(rowoff + r) * 72 + l15]      = f2bf(p0);
            plds[wv][(rowoff + r) * 72 + 16 + l15] = f2bf(p1);
        }

#pragma unroll
        for (int dn = 0; dn < 8; ++dn)
#pragma unroll
            for (int r = 0; r < 8; ++r) o[dn][r] *= ar[r];

        __builtin_amdgcn_wave_barrier();   // keep DS write->read ordered (in-order per wave)

        // ---- reload P as 16x32 A fragment ----
        FragBf pf;
        {
            const uint16_t* pp = &plds[wv][l15 * 72 + lh * 8];
            ld2x128(pf, pp, pp + 16);
        }

        // ---- O += P * V (V^T buffer gives contiguous 8-key runs for B fragments) ----
#pragma unroll
        for (int dn = 0; dn < 8; ++dn) {
            const uint16_t* vp = vh + (size_t)(16 * dn + l15) * T_SEQ + k0 + lh * 8;
            FragBf vf;
            ld2x128(vf, vp, vp + 16);
            o[dn] = wmma_bf16(pf.v, vf.v, o[dn]);
        }
    }

    // ---- normalize and store ----
#pragma unroll
    for (int r = 0; r < 8; ++r) lrow[r] = 1.f / lrow[r];
#pragma unroll
    for (int dn = 0; dn < 8; ++dn)
#pragma unroll
        for (int r = 0; r < 8; ++r)
            yb[(size_t)(q0 + rowoff + r) * D_MODEL + h * HEAD_DIM + 16 * dn + l15] =
                f2bf(o[dn][r] * lrow[r]);
}

// ---------------------------------------------------------------- launch
extern "C" void kernel_launch(void* const* d_in, const int* in_sizes, int n_in,
                              void* d_out, int out_size, void* d_ws, size_t ws_size,
                              hipStream_t stream) {
    (void)in_sizes; (void)n_in; (void)out_size; (void)ws_size;
    const float* x   = (const float*)d_in[0];
    const float* wq  = (const float*)d_in[1];
    const float* wk  = (const float*)d_in[2];
    const float* wv  = (const float*)d_in[3];
    const float* wo  = (const float*)d_in[4];
    const float* qkw = (const float*)d_in[5];

    char* ws = (char*)d_ws;                         // ~76 MB total, 16B-aligned offsets
    uint16_t* xb  = (uint16_t*)(ws + 0);            // 4096x2048 bf16
    uint16_t* wqb = (uint16_t*)(ws + 16777216);     // 2048x2048
    uint16_t* wkb = (uint16_t*)(ws + 25165824);     //  512x2048
    uint16_t* wvb = (uint16_t*)(ws + 27262976);     //  512x2048
    uint16_t* wob = (uint16_t*)(ws + 29360128);     // 2048x2048
    uint16_t* qb  = (uint16_t*)(ws + 37748736);     // 4096x2048 (Q, rope/norm in place)
    uint16_t* kb  = (uint16_t*)(ws + 54525952);     // 4096x512  (K, rope/norm in place)
    uint16_t* vtb = (uint16_t*)(ws + 58720256);     //  512x4096 (V transposed)
    uint16_t* yb  = (uint16_t*)(ws + 62914560);     // 4096x2048 (attention out)

    // bf16 staging
    f32_to_bf16_kernel<<<(8388608 + 255) / 256, 256, 0, stream>>>(x,  xb,  8388608);
    f32_to_bf16_kernel<<<(4194304 + 255) / 256, 256, 0, stream>>>(wq, wqb, 4194304);
    f32_to_bf16_kernel<<<(1048576 + 255) / 256, 256, 0, stream>>>(wk, wkb, 1048576);
    f32_to_bf16_kernel<<<(1048576 + 255) / 256, 256, 0, stream>>>(wv, wvb, 1048576);
    f32_to_bf16_kernel<<<(4194304 + 255) / 256, 256, 0, stream>>>(wo, wob, 4194304);

    // projections: waves = (M/64)*(N/64), 8 waves per 256-thread block
    gemm_wmma_kernel<0><<<256, 256, 0, stream>>>(xb, wqb, qb,  4096, 2048, 2048); // Q
    gemm_wmma_kernel<0><<< 64, 256, 0, stream>>>(xb, wkb, kb,  4096,  512, 2048); // K
    gemm_wmma_kernel<1><<< 64, 256, 0, stream>>>(xb, wvb, vtb, 4096,  512, 2048); // V (transposed)

    // RoPE + QK-norm, in place (one wave per 128-wide head row)
    rope_qknorm_kernel<<<8192, 256, 0, stream>>>(qb, qkw, N_HEADS);
    rope_qknorm_kernel<<<2048, 256, 0, stream>>>(kb, qkw, N_KV);

    // attention: 16 heads * 256 q-tiles = 4096 waves
    swa_attn_kernel<<<512, 256, 0, stream>>>(qb, kb, vtb, yb);

    // output projection -> f32
    gemm_wmma_kernel<2><<<256, 256, 0, stream>>>(yb, wob, d_out, 4096, 2048, 2048);
}
